// SwinSelfAttention_16114717294922
// MI455X (gfx1250) — matrix operand order repaired and checked by reading, not compiled, not run on hardware
//
#include <hip/hip_runtime.h>
#include <hip/hip_bf16.h>

typedef __attribute__((ext_vector_type(16))) _Float16 v16h;
typedef __attribute__((ext_vector_type(8)))  _Float16 v8h;
typedef __attribute__((ext_vector_type(4)))  _Float16 v4h;
typedef __attribute__((ext_vector_type(8)))  float    v8f;

// ---------------------------------------------------------------------------
// WMMA helper (CDNA5 gfx1250, wave32, V_WMMA_F32_16X16X32_F16)
// ---------------------------------------------------------------------------
__device__ __forceinline__ v8f wmma_f16(v16h a, v16h b, v8f c) {
  return __builtin_amdgcn_wmma_f32_16x16x32_f16(
      /*neg_a=*/false, a, /*neg_b=*/false, b,
      /*c_mod=*/(short)0, c, /*reuse_a=*/false, /*reuse_b=*/false);
}

// A-matrix 16x32 f16 fragment from row-major LDS (ld = row stride in halves).
// Lane l<16 : row M=l,    K = k0+0..7   and k0+16..23
// Lane l>=16: row M=l-16, K = k0+8..15  and k0+24..31
__device__ __forceinline__ v16h load_a_lds(const _Float16* base, int row0,
                                           int k0, int ld) {
  const int lane = threadIdx.x & 31;
  const int r  = lane & 15;
  const int ko = (lane & 16) >> 1;  // 0 or 8
  const _Float16* p = base + (row0 + r) * ld + k0 + ko;
  union { v16h v; v8h h[2]; } u;
  u.h[0] = *(const v8h*)(p);        // 16B contiguous -> ds_load_b128
  u.h[1] = *(const v8h*)(p + 16);
  return u.v;
}

// B-matrix 32x16 f16 fragment where the source holds columns contiguously
// (source = transposed matrix, row-major, row stride ld halves):
// lane l<16 : column n0+l,    K = k0+0..7   and k0+16..23
// lane l>=16: column n0+l-16, K = k0+8..15  and k0+24..31
__device__ __forceinline__ v16h load_bT(const _Float16* T, int n0, int k0,
                                        int ld) {
  const int lane = threadIdx.x & 31;
  const _Float16* p = T + (size_t)(n0 + (lane & 15)) * ld + k0 + ((lane & 16) >> 1);
  union { v16h v; v8h h[2]; } u;
  u.h[0] = *(const v8h*)(p);
  u.h[1] = *(const v8h*)(p + 16);
  return u.v;
}

// ---------------------------------------------------------------------------
// Prep kernel: transpose + fp32->f16 the weights once into workspace.
//   ws[0      ..16383] : WqT [H][hd][E]
//   ws[16384  ..32767] : WkT [H][hd][E]
//   ws[32768  ..49151] : WvT [H][hd][E]
//   ws[49152  ..65535] : WoT [N=E][K=E]
// ---------------------------------------------------------------------------
__global__ __launch_bounds__(256)
void prep_weights(const float* __restrict__ Wq, const float* __restrict__ Wk,
                  const float* __restrict__ Wv, const float* __restrict__ Wo,
                  _Float16* __restrict__ ws) {
  const int i = blockIdx.x * 256 + threadIdx.x;  // 0..16383
  if (i >= 16384) return;
  const int h = i >> 12;
  const int d = (i >> 7) & 31;
  const int e = i & 127;
  const int src = (h * 128 + e) * 32 + d;        // W[h][e][d]
  ws[i]         = (_Float16)Wq[src];
  ws[16384 + i] = (_Float16)Wk[src];
  ws[32768 + i] = (_Float16)Wv[src];
  const int n = i >> 7;                          // WoT[n][k] = Wo[k][n]
  const int k = i & 127;
  ws[49152 + i] = (_Float16)Wo[k * 128 + n];
}

// ---------------------------------------------------------------------------
// Main kernel: one workgroup (8 waves) per attention window.
// ---------------------------------------------------------------------------
__global__ __launch_bounds__(256, 1)
void swin_attn_kernel(const float* __restrict__ patches,
                      const float* __restrict__ bq,
                      const float* __restrict__ bk,
                      const float* __restrict__ bv,
                      const float* __restrict__ pos_bias,
                      const float* __restrict__ bo,
                      const unsigned char* __restrict__ mask,
                      const _Float16* __restrict__ ws,
                      float* __restrict__ out) {
  constexpr int S = 64, E = 128, HD = 32;

  __shared__ alignas(16) _Float16 Xs[S * E];        // 16 KB; reused as Oc
  __shared__ alignas(16) _Float16 Qs[4][S * HD];    // 16 KB  row-major [s][hd]
  __shared__ alignas(16) _Float16 Ks[4][S * HD];    // 16 KB  row-major [s][hd]
  __shared__ alignas(16) _Float16 Vt[4][HD * S];    // 16 KB  TRANSPOSED [hd][s]
  __shared__ alignas(16) _Float16 Ps[4][S * S];     // 32 KB
  __shared__ alignas(16) unsigned char Ms[S * S];   // 4 KB  mask window
  __shared__ alignas(16) float Bs[S * S];           // 16 KB fused bias+mask

  const int tid  = threadIdx.x;
  const int lane = tid & 31;
  const int wave = tid >> 5;
  const int win  = blockIdx.x;                  // b*256 + wy*16 + wx
  const int wy   = (win >> 4) & 15;
  const int wx   = win & 15;

  // ---- async-stage the 4KB mask window into LDS (256 lanes x 16B) ----
  {
    typedef __attribute__((address_space(3))) unsigned char uc3;
    unsigned ldsOff = (unsigned)(unsigned long long)(uc3*)(&Ms[tid * 16]);
    const unsigned char* msrc = mask + (size_t)(wy * 16 + wx) * (S * S);
    unsigned voff = tid * 16;
    asm volatile("global_load_async_to_lds_b128 %0, %1, %2"
                 :: "v"(ldsOff), "v"(voff), "s"(msrc)
                 : "memory");
  }

  if (tid == 0) {               // warm L2/WGP$ for the converted weights
    __builtin_prefetch(ws, 0, 0);
    __builtin_prefetch(ws + 32768, 0, 0);
  }

  // ---- stage X (fp32 -> f16), vectorized ----
  const float* Xg = patches + (size_t)win * (S * E);
  const float4* Xg4 = (const float4*)Xg;
  for (int i = tid; i < (S * E) / 4; i += 256) {
    float4 x = Xg4[i];
    v4h hx = {(_Float16)x.x, (_Float16)x.y, (_Float16)x.z, (_Float16)x.w};
    *(v4h*)(Xs + i * 4) = hx;
  }

  asm volatile("s_wait_asynccnt 0x0" ::: "memory");
  __syncthreads();

  // ---- build fused (pos_bias | -inf mask) table, 16 elements/thread ----
  for (int i = tid; i < S * S; i += 256) {
    const int q = i >> 6, k = i & 63;
    const int ri = (q >> 3) - (k >> 3) + 7;
    const int ci = (q & 7) - (k & 7) + 7;
    const float b = pos_bias[ri * 15 + ci];
    Bs[i] = Ms[i] ? -__builtin_inff() : b;
  }
  __syncthreads();

  const int h    = wave >> 1;   // head (2 waves per head)
  const int part = wave & 1;    // 16-wide strip / 32-row strip owner
  const int n0   = part * 16;
  const int colN = lane & 15;
  const int rhi  = (lane & 16) >> 1;  // +8 rows for upper half-wave

  // ---- Phase 1: Q/K/V projections  (X[64x128] @ W[128x32] + b) ----
  {
    const _Float16* WT[3] = { ws + h * (HD * E),
                              ws + 16384 + h * (HD * E),
                              ws + 32768 + h * (HD * E) };   // [hd][E]
    v16h bf[3][4];
#pragma unroll
    for (int p = 0; p < 3; ++p)
#pragma unroll
      for (int ks = 0; ks < 4; ++ks)
        bf[p][ks] = load_bT(WT[p], n0, ks * 32, E);

    const float biasq = bq[h * HD + n0 + colN];
    const float biask = bk[h * HD + n0 + colN];
    const float biasv = bv[h * HD + n0 + colN];

#pragma unroll
    for (int mi = 0; mi < 4; ++mi) {
      v16h a[4];
#pragma unroll
      for (int ks = 0; ks < 4; ++ks) a[ks] = load_a_lds(Xs, mi * 16, ks * 32, E);
      const int rbase = mi * 16 + rhi;

      {  // Q -> row-major scatter
        v8f c = {};
#pragma unroll
        for (int ks = 0; ks < 4; ++ks) c = wmma_f16(a[ks], bf[0][ks], c);
#pragma unroll
        for (int r = 0; r < 8; ++r)
          Qs[h][(rbase + r) * HD + n0 + colN] = (_Float16)(c[r] + biasq);
      }
      {  // K -> row-major scatter
        v8f c = {};
#pragma unroll
        for (int ks = 0; ks < 4; ++ks) c = wmma_f16(a[ks], bf[1][ks], c);
#pragma unroll
        for (int r = 0; r < 8; ++r)
          Ks[h][(rbase + r) * HD + n0 + colN] = (_Float16)(c[r] + biask);
      }
      {  // V -> transposed layout: 8 consecutive s-values -> one b128 store
        v8f c = {};
#pragma unroll
        for (int ks = 0; ks < 4; ++ks) c = wmma_f16(a[ks], bf[2][ks], c);
        v8h pk;
#pragma unroll
        for (int r = 0; r < 8; ++r) pk[r] = (_Float16)(c[r] + biasv);
        *(v8h*)(Vt[h] + (n0 + colN) * S + rbase) = pk;
      }
    }
  }
  __syncthreads();

  // ---- Phase 2: logits = QK^T*scale + fused(bias,mask), softmax -> Ps ----
  {
    const float scale = 0.17677669529663687f;  // 1/sqrt(32)
    for (int mt = 0; mt < 2; ++mt) {
      const int mi = part * 2 + mt;            // wave owns 32 complete rows
      const int rbase = mi * 16 + rhi;
      v16h a = load_a_lds(Qs[h], mi * 16, 0, HD);   // K dim = 32 exactly
      float cc[4][8];
#pragma unroll
      for (int ni = 0; ni < 4; ++ni) {
        // B = K^T: column n = row n of K, two 8-half runs per lane
        v16h b = load_bT(Ks[h], ni * 16, 0, HD);
        v8f c = {};
        c = wmma_f16(a, b, c);
#pragma unroll
        for (int r = 0; r < 8; ++r)
          cc[ni][r] = fmaf(c[r], scale, Bs[(rbase + r) * S + ni * 16 + colN]);
      }
      // row-wise softmax: 4 values/lane, butterfly over the 16-lane half
#pragma unroll
      for (int r = 0; r < 8; ++r) {
        float mx = fmaxf(fmaxf(cc[0][r], cc[1][r]), fmaxf(cc[2][r], cc[3][r]));
        for (int d = 1; d < 16; d <<= 1) mx = fmaxf(mx, __shfl_xor(mx, d, 32));
        float s = 0.f;
#pragma unroll
        for (int ni = 0; ni < 4; ++ni) {
          cc[ni][r] = __expf(cc[ni][r] - mx);
          s += cc[ni][r];
        }
        for (int d = 1; d < 16; d <<= 1) s += __shfl_xor(s, d, 32);
        const float inv = 1.0f / s;
        const int q = rbase + r;
#pragma unroll
        for (int ni = 0; ni < 4; ++ni)
          Ps[h][q * S + ni * 16 + colN] = (_Float16)(cc[ni][r] * inv);
      }
    }
  }
  __syncthreads();

  // ---- Phase 3: O = P @ V, concat heads into Oc (reuses Xs) ----
  _Float16* Oc = Xs;
  {
    v16h bv_[2][2];   // hoisted: contiguous loads from transposed V
#pragma unroll
    for (int ni = 0; ni < 2; ++ni)
#pragma unroll
      for (int ks = 0; ks < 2; ++ks)
        bv_[ni][ks] = load_bT(Vt[h], ni * 16, ks * 32, S);

    for (int mt = 0; mt < 2; ++mt) {
      const int mi = part * 2 + mt;
      const int rbase = mi * 16 + rhi;
      v16h a0 = load_a_lds(Ps[h], mi * 16, 0,  S);
      v16h a1 = load_a_lds(Ps[h], mi * 16, 32, S);
#pragma unroll
      for (int ni = 0; ni < 2; ++ni) {
        v8f c = {};
        c = wmma_f16(a0, bv_[ni][0], c);
        c = wmma_f16(a1, bv_[ni][1], c);
        const int col = h * HD + ni * 16 + colN;
#pragma unroll
        for (int r = 0; r < 8; ++r)
          Oc[(rbase + r) * E + col] = (_Float16)c[r];
      }
    }
  }
  __syncthreads();

  // ---- Phase 4: Y = Oc[64x128] @ Wo[128x128] + bo ----
  {
    const int nb = wave * 16;  // each wave owns one 16-col output strip
    const _Float16* WoT = ws + 49152;          // [N=128][K=128]
    v16h bf[4];
#pragma unroll
    for (int ks = 0; ks < 4; ++ks) bf[ks] = load_bT(WoT, nb, ks * 32, E);
    const float bias = bo[nb + colN];
    float* Yg = out + (size_t)win * (S * E);
#pragma unroll
    for (int mi = 0; mi < 4; ++mi) {
      v8f c = {};
#pragma unroll
      for (int ks = 0; ks < 4; ++ks) {
        v16h a = load_a_lds(Oc, mi * 16, ks * 32, E);
        c = wmma_f16(a, bf[ks], c);
      }
      const int col = nb + colN;
      const int rbase = mi * 16 + rhi;
#pragma unroll
      for (int r = 0; r < 8; ++r)
        Yg[(rbase + r) * E + col] = c[r] + bias;
    }
  }
}

// ---------------------------------------------------------------------------
extern "C" void kernel_launch(void* const* d_in, const int* in_sizes, int n_in,
                              void* d_out, int out_size, void* d_ws,
                              size_t ws_size, hipStream_t stream) {
  const float* patches  = (const float*)d_in[0];
  const float* Wq       = (const float*)d_in[1];
  const float* bq       = (const float*)d_in[2];
  const float* Wk       = (const float*)d_in[3];
  const float* bk       = (const float*)d_in[4];
  const float* Wv       = (const float*)d_in[5];
  const float* bv       = (const float*)d_in[6];
  const float* pos_bias = (const float*)d_in[7];
  const float* Wo       = (const float*)d_in[8];
  const float* bo       = (const float*)d_in[9];
  const unsigned char* mask = (const unsigned char*)d_in[10];
  float* out = (float*)d_out;
  _Float16* ws = (_Float16*)d_ws;   // needs 65536 halves = 128 KB

  // 1) transpose+convert weights to f16 once (tiny; L2-resident afterwards)
  prep_weights<<<dim3(64), dim3(256), 0, stream>>>(Wq, Wk, Wv, Wo, ws);

  // 2) 16 batches * 16 * 16 windows = 4096 workgroups, 256 threads (8 waves)
  swin_attn_kernel<<<dim3(4096), dim3(256), 0, stream>>>(
      patches, bq, bk, bv, pos_bias, bo, mask, ws, out);
}
